// GNNBlock_57604101374372
// MI455X (gfx1250) — matrix operand order repaired and checked by reading, not compiled, not run on hardware
//
#include <hip/hip_runtime.h>
#include <hip/hip_bf16.h>

typedef _Float16 v8h  __attribute__((ext_vector_type(8)));
typedef _Float16 v16h __attribute__((ext_vector_type(16)));
typedef float    v8f  __attribute__((ext_vector_type(8)));

#define ALPHA 0.2f
#define EPSN  1e-5f

__device__ __forceinline__ float lrelu(float x){ return x > 0.f ? x : ALPHA * x; }

// Order-preserving float -> uint encoding (for atomicMax on f32 scores)
__device__ __forceinline__ unsigned fenc(float f){
  unsigned u = __float_as_uint(f);
  return (u & 0x80000000u) ? ~u : (u | 0x80000000u);
}
__device__ __forceinline__ float fdec(unsigned k){
  return __uint_as_float((k & 0x80000000u) ? (k ^ 0x80000000u) : ~k);
}

// ---------------------------------------------------------------- utilities
__global__ void k_zero(float* p, int n){
  int i = blockIdx.x * blockDim.x + threadIdx.x;
  if (i < n) p[i] = 0.f;
}

__global__ void k_cvt(const float* __restrict__ x, _Float16* __restrict__ xh, int n){
  int i = blockIdx.x * blockDim.x + threadIdx.x;
  if (i < n) xh[i] = (_Float16)x[i];
}

// Pack fp32 weights [K=128][N=128] row-major into this file's contiguous-k (not AMD table order) wave32 WMMA
// B-matrix layout for V_WMMA_F32_16X16X32_F16:
//   per 32x16 (KxN) tile: lane = (k%32/16)*16 + n%16 ; element e = k%16
// so each lane's 16 halves are contiguous -> two GLOBAL_LOAD_B128 per fragment.
__global__ void k_pack(const float* __restrict__ Wq, const float* __restrict__ Wk,
                       const float* __restrict__ Wv, _Float16* __restrict__ pw, int nmat){
  int i = blockIdx.x * blockDim.x + threadIdx.x;
  int total = nmat * 3 * 128 * 128;
  if (i >= total) return;
  int r     = i & 16383;   // element within one 128x128 matrix (k*128+n)
  int m     = i >> 14;     // matrix id = (layerbranch*3 + which)
  int which = m % 3;
  int lb    = m / 3;
  int kk = r >> 7, n = r & 127;
  const float* src = (which == 0) ? Wq : ((which == 1) ? Wk : Wv);
  float val = src[(size_t)lb * 16384 + r];
  int kt = kk >> 5, k32 = kk & 31;
  int half = k32 >> 4, e = k32 & 15;
  int lane = half * 16 + (n & 15);
  int nt   = n >> 4;
  pw[(size_t)m * 16384 + (((nt * 4 + kt) * 32 + lane) << 4) + e] = (_Float16)val;
}

// ------------------------------------------------- fused q/k/v GEMM (WMMA)
// Block: 256 threads = 8 waves. Each wave owns a 32x16 output tile (two
// 16x16 M-tiles sharing every B fragment); wave w -> column tile nt=w, so a
// block covers a 32-row x 128-col stripe. 24 WMMAs per wave.
// q,k get bias+leaky_relu; v gets bias only.
__global__ void __launch_bounds__(256) k_qkv(
    const _Float16* __restrict__ xh, const _Float16* __restrict__ pw,
    const float* __restrict__ bq, const float* __restrict__ bk, const float* __restrict__ bv,
    float* __restrict__ q, float* __restrict__ k, float* __restrict__ v, int N)
{
  int wave = threadIdx.x >> 5, lane = threadIdx.x & 31;
  int row0 = blockIdx.x * 32;
  int nt = wave;
  int ma = lane & 15, h = lane >> 4;
  const _Float16* arow0 = xh + (size_t)(row0 + ma) * 128;
  const _Float16* arow1 = arow0 + 16 * 128;

  v8f cq0 = {}, ck0 = {}, cv0 = {};
  v8f cq1 = {}, ck1 = {}, cv1 = {};
  union AB { v16h v; v8h h2[2]; };

  #pragma unroll
  for (int kt = 0; kt < 4; ++kt){
    AB a0, a1;
    // A operand k order used here (not AMD's table order): lane half h, element e -> k = h*16+e (matches packed B)
    a0.h2[0] = *(const v8h*)(arow0 + kt * 32 + h * 16);
    a0.h2[1] = *(const v8h*)(arow0 + kt * 32 + h * 16 + 8);
    a1.h2[0] = *(const v8h*)(arow1 + kt * 32 + h * 16);
    a1.h2[1] = *(const v8h*)(arow1 + kt * 32 + h * 16 + 8);
    int boff = ((nt * 4 + kt) * 32 + lane) << 4;
    AB b0, b1, b2;
    b0.h2[0] = *(const v8h*)(pw + boff);
    b0.h2[1] = *(const v8h*)(pw + boff + 8);
    b1.h2[0] = *(const v8h*)(pw + 16384 + boff);
    b1.h2[1] = *(const v8h*)(pw + 16384 + boff + 8);
    b2.h2[0] = *(const v8h*)(pw + 32768 + boff);
    b2.h2[1] = *(const v8h*)(pw + 32768 + boff + 8);
    cq0 = __builtin_amdgcn_wmma_f32_16x16x32_f16(false, a0.v, false, b0.v, (short)0, cq0, false, false);
    cq1 = __builtin_amdgcn_wmma_f32_16x16x32_f16(false, a1.v, false, b0.v, (short)0, cq1, false, false);
    ck0 = __builtin_amdgcn_wmma_f32_16x16x32_f16(false, a0.v, false, b1.v, (short)0, ck0, false, false);
    ck1 = __builtin_amdgcn_wmma_f32_16x16x32_f16(false, a1.v, false, b1.v, (short)0, ck1, false, false);
    cv0 = __builtin_amdgcn_wmma_f32_16x16x32_f16(false, a0.v, false, b2.v, (short)0, cv0, false, false);
    cv1 = __builtin_amdgcn_wmma_f32_16x16x32_f16(false, a1.v, false, b2.v, (short)0, cv1, false, false);
  }

  // C/D layout: VGPR r, lane l -> M = r + (l/16)*8, N = l%16
  int n = lane & 15, hh = lane >> 4;
  float bqv = bq[nt * 16 + n], bkv = bk[nt * 16 + n], bvv = bv[nt * 16 + n];
  #pragma unroll
  for (int r = 0; r < 8; ++r){
    size_t o0 = (size_t)(row0 + hh * 8 + r) * 128 + nt * 16 + n;
    size_t o1 = o0 + (size_t)16 * 128;
    q[o0] = lrelu(cq0[r] + bqv);
    k[o0] = lrelu(ck0[r] + bkv);
    v[o0] = cv0[r] + bvv;
    q[o1] = lrelu(cq1[r] + bqv);
    k[o1] = lrelu(ck1[r] + bkv);
    v[o1] = cv1[r] + bvv;
  }
}

// --------------------------------------------------------- edge kernels
// One wave per edge: wid is wave-uniform, so indices are scalarized with
// readfirstlane -> row bases live in SGPRs (saddr-form loads).
__global__ void k_edge_score(const float* __restrict__ q, const float* __restrict__ k,
    const int* __restrict__ erow, const int* __restrict__ ecol,
    float* __restrict__ sc, unsigned* __restrict__ smax, int E, int Etot)
{
  int wid  = (int)((blockIdx.x * blockDim.x + threadIdx.x) >> 5);
  int lane = threadIdx.x & 31;
  if (wid >= Etot) return;
  int r = (wid < E) ? erow[wid] : (wid - E);   // self-loops appended
  int c = (wid < E) ? ecol[wid] : (wid - E);
  r = __builtin_amdgcn_readfirstlane(r);
  c = __builtin_amdgcn_readfirstlane(c);
  const float4 qv = *(const float4*)(q + (size_t)r * 128 + lane * 4);
  const float4 kv = *(const float4*)(k + (size_t)c * 128 + lane * 4);
  float s = qv.x * kv.x + qv.y * kv.y + qv.z * kv.z + qv.w * kv.w;
  #pragma unroll
  for (int o = 16; o > 0; o >>= 1) s += __shfl_xor(s, o, 32);
  if (lane == 0){
    sc[wid] = s;
    atomicMax(&smax[r], fenc(s));
  }
}

__global__ void k_edge_exp(float* __restrict__ sc, const unsigned* __restrict__ smax,
    float* __restrict__ denom, const int* __restrict__ erow, int E, int Etot)
{
  int i = blockIdx.x * blockDim.x + threadIdx.x;
  if (i >= Etot) return;
  int r = (i < E) ? erow[i] : (i - E);
  float ex = expf(sc[i] - fdec(smax[r]));
  sc[i] = ex;
  atomicAdd(&denom[r], ex);
}

__global__ void k_edge_agg(const float* __restrict__ sc, const float* __restrict__ denom,
    const float* __restrict__ v, const int* __restrict__ erow, const int* __restrict__ ecol,
    float* __restrict__ agg, int E, int Etot)
{
  int wid  = (int)((blockIdx.x * blockDim.x + threadIdx.x) >> 5);
  int lane = threadIdx.x & 31;
  if (wid >= Etot) return;
  int r = (wid < E) ? erow[wid] : (wid - E);
  int c = (wid < E) ? ecol[wid] : (wid - E);
  r = __builtin_amdgcn_readfirstlane(r);
  c = __builtin_amdgcn_readfirstlane(c);
  float att = sc[wid] / denom[r];
  const float4 vv = *(const float4*)(v + (size_t)c * 128 + lane * 4);
  float* dst = agg + (size_t)r * 128 + lane * 4;
  atomicAdd(dst + 0, att * vv.x);
  atomicAdd(dst + 1, att * vv.y);
  atomicAdd(dst + 2, att * vv.z);
  atomicAdd(dst + 3, att * vv.w);
}

// ------------------------------------------------------ graph norm
// Block = 128 threads (one per feature u), block covers (graph b, 64-node chunk).
__global__ void k_stats(const float* __restrict__ agg, float* __restrict__ sums,
                        float* __restrict__ sumsq, int F, int chunks)
{
  int u  = threadIdx.x;
  int b  = blockIdx.x / chunks;
  int ch = blockIdx.x % chunks;
  int f0 = ch * 64;
  float s = 0.f, s2 = 0.f;
  for (int f = f0; f < f0 + 64 && f < F; ++f){
    float x = agg[((size_t)b * F + f) * 128 + u];
    s += x; s2 += x * x;
  }
  atomicAdd(&sums[b * 128 + u], s);
  atomicAdd(&sumsq[b * 128 + u], s2);
}

__global__ void k_apply(const float* __restrict__ agg, const float* __restrict__ sums,
    const float* __restrict__ sumsq, const float* __restrict__ gamma,
    const float* __restrict__ beta, float* __restrict__ out, _Float16* __restrict__ xh,
    int last, int N, int F, float invF)
{
  int i = blockIdx.x * blockDim.x + threadIdx.x;
  if (i >= N * 128) return;
  int u = i & 127, node = i >> 7;
  int b = node / F, f = node % F;
  float mean = sums[b * 128 + u] * invF;
  float var  = sumsq[b * 128 + u] * invF - mean * mean;
  float sd   = sqrtf(fmaxf(var, 0.f));
  float xn   = (agg[i] - mean) / (sd + EPSN);
  float y    = lrelu(gamma[(size_t)f * 128 + u] * xn + beta[(size_t)f * 128 + u]);
  if (last) out[i] = y;
  else      xh[i]  = (_Float16)y;
}

// ---------------------------------------------------------------- driver
extern "C" void kernel_launch(void* const* d_in, const int* in_sizes, int n_in,
                              void* d_out, int out_size, void* d_ws, size_t ws_size,
                              hipStream_t stream)
{
  const int U = 128;
  const int N = in_sizes[0] / U;
  const int E = in_sizes[1] / 2;
  const int L = in_sizes[2] / (2 * U * U);
  const int F = in_sizes[8] / (L * U);
  const int B = N / F;
  const int Etot = E + N;

  const float* g     = (const float*)d_in[0];
  const int*   e     = (const int*)  d_in[1];
  const float* Wq    = (const float*)d_in[2];
  const float* bq    = (const float*)d_in[3];
  const float* Wk    = (const float*)d_in[4];
  const float* bk    = (const float*)d_in[5];
  const float* Wv    = (const float*)d_in[6];
  const float* bv    = (const float*)d_in[7];
  const float* gamma = (const float*)d_in[8];
  const float* beta  = (const float*)d_in[9];
  float* out = (float*)d_out;

  char* ws = (char*)d_ws;
  size_t off = 0;
  auto carve = [&](size_t bytes) -> char* {
    char* p = ws + off;
    off = (off + bytes + 255) & ~(size_t)255;
    return p;
  };
  _Float16* pw   = (_Float16*)carve((size_t)L * 2 * 3 * 16384 * 2);
  _Float16* xh   = (_Float16*)carve((size_t)N * U * 2);
  float*    q    = (float*)carve((size_t)N * U * 4);
  float*    kbuf = (float*)carve((size_t)N * U * 4);
  float*    v    = (float*)carve((size_t)N * U * 4);
  float*    agg  = (float*)carve((size_t)N * U * 4);
  float*    sc   = (float*)carve((size_t)Etot * 4);
  unsigned* smax = (unsigned*)carve((size_t)N * 4 * 2);  // smax | denom contiguous
  float*    den  = (float*)(smax + N);
  float*    sums = (float*)carve((size_t)B * U * 4 * 2);
  float*    sumsq = sums + (size_t)B * U;

  {
    int tot = L * 2 * 3 * 16384;
    k_pack<<<(tot + 255) / 256, 256, 0, stream>>>(Wq, Wk, Wv, pw, L * 2);
  }
  k_cvt<<<(N * U + 255) / 256, 256, 0, stream>>>(g, xh, N * U);

  const int* erow = e;
  const int* ecol = e + E;
  const int wblocks = (Etot + 7) / 8;   // 8 waves per 256-thread block

  for (int l = 0; l < L; ++l){
    k_zero<<<(N * U + 255) / 256, 256, 0, stream>>>(agg, N * U);
    for (int br = 0; br < 2; ++br){
      int lb = l * 2 + br;
      k_zero<<<(2 * N + 255) / 256, 256, 0, stream>>>((float*)smax, 2 * N);
      k_qkv<<<N / 32, 256, 0, stream>>>(xh, pw + (size_t)lb * 3 * 16384,
          bq + (size_t)lb * U, bk + (size_t)lb * U, bv + (size_t)lb * U,
          q, kbuf, v, N);
      k_edge_score<<<wblocks, 256, 0, stream>>>(q, kbuf, erow, ecol, sc, smax, E, Etot);
      k_edge_exp<<<(Etot + 255) / 256, 256, 0, stream>>>(sc, smax, den, erow, E, Etot);
      k_edge_agg<<<wblocks, 256, 0, stream>>>(sc, den, v, erow, ecol, agg, E, Etot);
    }
    k_zero<<<(2 * B * U + 255) / 256, 256, 0, stream>>>(sums, 2 * B * U);
    int chunks = (F + 63) / 64;
    k_stats<<<B * chunks, 128, 0, stream>>>(agg, sums, sumsq, F, chunks);
    k_apply<<<(N * U + 255) / 256, 256, 0, stream>>>(agg, sums, sumsq,
        gamma + (size_t)l * F * U, beta + (size_t)l * F * U,
        out, xh, (l == L - 1) ? 1 : 0, N, F, 1.0f / (float)F);
  }
}
